// ConstrainedRNN_72292889526986
// MI455X (gfx1250) — matrix operand order repaired and checked by reading, not compiled, not run on hardware
//
#include <hip/hip_runtime.h>
#include <cmath>

// ---------------------------------------------------------------------------
// Leaky RNN (ConstrainedRNN): B=64, T=2048, I=128, H=512, O=64, alpha=0.2
//   xp  = X @ W_in^T + b_in                     (parallel GEMM, WMMA f16)
//   h_t = 0.8*h + 0.2*tanh(xp_t + h @ W_rec^T + h_bias)   (sequential scan,
//         async-to-LDS double-buffered xp prefetch under the WMMA shadow)
//   out = hs @ W_out^T + b_out                  (parallel GEMM, WMMA f16)
// ---------------------------------------------------------------------------

typedef _Float16 half8  __attribute__((ext_vector_type(8)));
typedef _Float16 v16h   __attribute__((ext_vector_type(16)));
typedef float    v8f    __attribute__((ext_vector_type(8)));
typedef int      v4i    __attribute__((ext_vector_type(4)));

static constexpr int B_ = 64, T_ = 2048, I_ = 128, H_ = 512, O_ = 64;
static constexpr float ALPHA_ = 0.2f;

#define AS1 __attribute__((address_space(1)))
#define AS3 __attribute__((address_space(3)))

#if __has_builtin(__builtin_amdgcn_global_load_async_to_lds_b128)
#define HAVE_ASYNC_LDS 1
// builtin expects int4* operands (per compiler diagnostic)
#define ASYNC_CP128(gp, lp) \
    __builtin_amdgcn_global_load_async_to_lds_b128( \
        (AS1 v4i*)(gp), (AS3 v4i*)(lp), 0, 0)
#else
#define HAVE_ASYNC_LDS 0
#endif

#if __has_builtin(__builtin_amdgcn_s_wait_asynccnt)
#define WAIT_ASYNCCNT(n) __builtin_amdgcn_s_wait_asynccnt(n)
#else
#define WAIT_ASYNCCNT(n) asm volatile("s_wait_asynccnt %0" :: "n"(n) : "memory")
#endif

__device__ __forceinline__ v16h combine16(half8 lo, half8 hi) {
    return __builtin_shufflevector(lo, hi, 0,1,2,3,4,5,6,7,8,9,10,11,12,13,14,15);
}
__device__ __forceinline__ v8f wmma_f16(v16h a, v16h b, v8f c) {
    return __builtin_amdgcn_wmma_f32_16x16x32_f16(false, a, false, b, (short)0, c,
                                                  false, false);
}

// ---------------------------------------------------------------------------
// Pack a weight matrix W [N][K] (row-major f32) into the WMMA B-matrix lane
// layout, f16. Blocks of 32(K) x 16(N); block id = nb*(K/32) + kb; within a
// block, lane L stores 16 contiguous f16: column n = nb*16 + L%16, rows
// k = kb*32 + (L/16)*8 + i for i < 8, k = kb*32 + 16 + (L/16)*8 + (i-8) for i >= 8.
// ---------------------------------------------------------------------------
__global__ void pack_b_kernel(const float* __restrict__ W,
                              _Float16* __restrict__ dst, int K, int N) {
    int e = blockIdx.x * blockDim.x + threadIdx.x;
    if (e >= K * N) return;
    int block  = e >> 9;
    int within = e & 511;
    int lane   = within >> 4;
    int i      = within & 15;
    int kblocks = K >> 5;
    int kb = block % kblocks;
    int nb = block / kblocks;
    int k = kb * 32 + (lane >> 4) * 8 + (i & 7) + (i >> 3) * 16;
    int n = nb * 16 + (lane & 15);
    dst[e] = (_Float16)W[(size_t)n * K + k];
}

// ---------------------------------------------------------------------------
// Phase 1: xp[m][h] = X_row(m) . W_in[h][:] + b_in[h], m = t*64 + b
// ---------------------------------------------------------------------------
__global__ __launch_bounds__(256) void xproj_kernel(
        const float* __restrict__ x, const _Float16* __restrict__ Wp,
        const float* __restrict__ b_in, float* __restrict__ xp) {
    int lane = threadIdx.x & 31;
    int wid  = (blockIdx.x * blockDim.x + threadIdx.x) >> 5;
    int mt = wid >> 5;          // 0..8191  (M/16)
    int nt = wid & 31;          // 0..31    (H/16)
    int hf = lane >> 4, lm = lane & 15;

    int m = mt * 16 + lm;       // m = t*64 + b
    int t = m >> 6, b = m & 63;
    const float* arow = x + ((size_t)b * T_ + t) * I_;

    v8f acc = {};
    #pragma unroll
    for (int kb = 0; kb < 4; ++kb) {
        const float* p = arow + kb * 32 + hf * 8;
        half8 lo, hi;
        #pragma unroll
        for (int j = 0; j < 8; ++j) {
            lo[j] = (_Float16)p[j];
            hi[j] = (_Float16)p[16 + j];
        }
        v16h a = combine16(lo, hi);
        v16h bf = *(const v16h*)(Wp + ((size_t)(nt * 4 + kb) * 512 + lane * 16));
        acc = wmma_f16(a, bf, acc);
    }
    int col = nt * 16 + lm;
    float bias = b_in[col];
    #pragma unroll
    for (int r = 0; r < 8; ++r) {
        int br = r + hf * 8;
        xp[((size_t)mt * 16 + br) * H_ + col] = acc[r] + bias;
    }
}

// ---------------------------------------------------------------------------
// Phase 2: sequential leaky-tanh scan. grid = 4 workgroups (16 batch rows
// each), 512 threads = 16 waves; wave w owns N-tiles 2w, 2w+1.
// h (16x512 f16) in LDS; xp[t] double-buffered in LDS via async-to-LDS
// prefetch issued before the WMMA k-loop; W_rec re-streamed from L2.
// ---------------------------------------------------------------------------
__global__ __launch_bounds__(512) void rnn_scan_kernel(
        const float* __restrict__ xp, const _Float16* __restrict__ Wrec_p,
        const float* __restrict__ h_bias, _Float16* __restrict__ hs) {
    __shared__ _Float16 h_lds[16 * 512];           // 16 KB
#if HAVE_ASYNC_LDS
    __shared__ float xp_buf[2][16 * 512];          // 2 x 32 KB
#endif
    int tid  = threadIdx.x;
    int lane = tid & 31;
    int w    = tid >> 5;                 // 0..15
    int hf = lane >> 4, lm = lane & 15;
    int bg = blockIdx.x;                 // batch group, rows bg*16..+15

    for (int i = tid; i < 16 * 512; i += 512) h_lds[i] = (_Float16)0.f;

    const float* xp_base = xp + (size_t)bg * 16 * H_;   // + t*64*H_ per step

#if HAVE_ASYNC_LDS
    // prefetch xp(0) into buf0: 16x512 f32 = 32 KB, 4 x b128 per thread
    {
        const float* g = xp_base + tid * 4;
        float* l = &xp_buf[0][tid * 4];
        #pragma unroll
        for (int i = 0; i < 4; ++i)
            ASYNC_CP128(g + i * 2048, l + i * 2048);
    }
#endif
    __syncthreads();

    int nb0 = w * 2, nb1 = w * 2 + 1;
    const _Float16* Bp0 = Wrec_p + (size_t)nb0 * 16 * 512 + lane * 16;
    const _Float16* Bp1 = Wrec_p + (size_t)nb1 * 16 * 512 + lane * 16;
    int col0 = nb0 * 16 + lm, col1 = nb1 * 16 + lm;
    float bias0 = h_bias[col0], bias1 = h_bias[col1];

    for (int t = 0; t < T_; ++t) {
#if HAVE_ASYNC_LDS
        // issue async prefetch of xp(t+1) before the WMMA chain
        if (t + 1 < T_) {
            const float* g = xp_base + (size_t)(t + 1) * 64 * H_ + tid * 4;
            float* l = &xp_buf[(t + 1) & 1][tid * 4];
            #pragma unroll
            for (int i = 0; i < 4; ++i)
                ASYNC_CP128(g + i * 2048, l + i * 2048);
        }
#endif
        v8f acc0 = {}, acc1 = {};
        #pragma unroll 4
        for (int kb = 0; kb < 16; ++kb) {
            const _Float16* ap = h_lds + lm * 512 + kb * 32 + hf * 8;
            half8 lo = *(const half8*)(ap);
            half8 hi = *(const half8*)(ap + 16);
            v16h a  = combine16(lo, hi);
            v16h b0 = *(const v16h*)(Bp0 + (size_t)kb * 512);
            v16h b1 = *(const v16h*)(Bp1 + (size_t)kb * 512);
            acc0 = wmma_f16(a, b0, acc0);
            acc1 = wmma_f16(a, b1, acc1);
        }
        // read h_old (all LDS h reads must precede the barrier)
        float ho0[8], ho1[8];
        #pragma unroll
        for (int r = 0; r < 8; ++r) {
            int br = r + hf * 8;
            ho0[r] = (float)h_lds[br * 512 + col0];
            ho1[r] = (float)h_lds[br * 512 + col1];
        }
#if HAVE_ASYNC_LDS
        // async loads complete in order: with the 4 just-issued ops
        // outstanding, <=4 guarantees step-t's slab has landed (this wave);
        // the barrier below makes it workgroup-wide.
        if (t + 1 < T_) { WAIT_ASYNCCNT(4); } else { WAIT_ASYNCCNT(0); }
#endif
        __syncthreads();

        float hn0[8], hn1[8];
        #pragma unroll
        for (int r = 0; r < 8; ++r) {
            int br = r + hf * 8;
#if HAVE_ASYNC_LDS
            float x0 = xp_buf[t & 1][br * 512 + col0];
            float x1 = xp_buf[t & 1][br * 512 + col1];
#else
            const float* xprow = xp_base + (size_t)t * 64 * H_;
            float x0 = xprow[(size_t)br * H_ + col0];
            float x1 = xprow[(size_t)br * H_ + col1];
#endif
            hn0[r] = (1.f - ALPHA_) * ho0[r] + ALPHA_ * tanhf(acc0[r] + x0 + bias0);
            hn1[r] = (1.f - ALPHA_) * ho1[r] + ALPHA_ * tanhf(acc1[r] + x1 + bias1);
        }
        #pragma unroll
        for (int r = 0; r < 8; ++r) {
            int br = r + hf * 8;
            h_lds[br * 512 + col0] = (_Float16)hn0[r];
            h_lds[br * 512 + col1] = (_Float16)hn1[r];
        }
        __syncthreads();
        // contiguous copy-out of h to hs[t][bg*16..+15][:] (f16)
        {
            _Float16* orow = hs + ((size_t)t * 64 + bg * 16) * H_;
            const half8* src = (const half8*)h_lds;
            half8* dstp = (half8*)orow;
            dstp[tid]       = src[tid];
            dstp[tid + 512] = src[tid + 512];
        }
        // next iteration's LDS writes are fenced by its own __syncthreads()
    }
}

// ---------------------------------------------------------------------------
// Phase 3: out[b][t][o] = hs_row(t*64+b) . W_out[o][:] + b_out[o]
// ---------------------------------------------------------------------------
__global__ __launch_bounds__(256) void out_kernel(
        const _Float16* __restrict__ hs, const _Float16* __restrict__ Wp,
        const float* __restrict__ b_out, float* __restrict__ out) {
    int lane = threadIdx.x & 31;
    int wid  = (blockIdx.x * blockDim.x + threadIdx.x) >> 5;
    int mt = wid >> 2;          // 0..8191
    int nt = wid & 3;           // 0..3  (O/16)
    int hf = lane >> 4, lm = lane & 15;

    const _Float16* arow = hs + (size_t)(mt * 16 + lm) * H_;
    v8f acc = {};
    #pragma unroll 4
    for (int kb = 0; kb < 16; ++kb) {
        const _Float16* p = arow + kb * 32 + hf * 8;
        half8 lo = *(const half8*)p;
        half8 hi = *(const half8*)(p + 16);
        v16h a = combine16(lo, hi);
        v16h b = *(const v16h*)(Wp + ((size_t)(nt * 16 + kb) * 512 + lane * 16));
        acc = wmma_f16(a, b, acc);
    }
    int o = nt * 16 + lm;
    float bias = b_out[o];
    #pragma unroll
    for (int r = 0; r < 8; ++r) {
        int br = r + hf * 8;
        int m  = mt * 16 + br;          // m = t*64 + b
        int t  = m >> 6, b = m & 63;
        out[((size_t)b * T_ + t) * O_ + o] = acc[r] + bias;
    }
}

// ---------------------------------------------------------------------------
extern "C" void kernel_launch(void* const* d_in, const int* in_sizes, int n_in,
                              void* d_out, int out_size, void* d_ws, size_t ws_size,
                              hipStream_t stream) {
    (void)in_sizes; (void)n_in; (void)out_size; (void)ws_size;
    const float* inputs = (const float*)d_in[0];   // [B,T,I]
    const float* W_in   = (const float*)d_in[1];   // [H,I]
    const float* b_in   = (const float*)d_in[2];   // [H]
    const float* W_rec  = (const float*)d_in[3];   // [H,H]
    const float* h_bias = (const float*)d_in[4];   // [H]
    const float* W_out  = (const float*)d_in[5];   // [O,H]
    const float* b_out  = (const float*)d_in[6];   // [O]
    float* out = (float*)d_out;                    // [B,T,O]

    // workspace layout (~385 MB)
    char* ws = (char*)d_ws;
    float*    xp   = (float*)ws;                                    // 256 MB
    _Float16* hs   = (_Float16*)(ws + (size_t)268435456);           // 128 MB
    _Float16* WinP = (_Float16*)(ws + (size_t)402653184);           // 128 KB
    _Float16* WrecP= (_Float16*)(ws + (size_t)402784256);           // 512 KB
    _Float16* WoutP= (_Float16*)(ws + (size_t)403308544);           // 64 KB

    pack_b_kernel<<<(H_ * I_ + 255) / 256, 256, 0, stream>>>(W_in,  WinP,  I_, H_);
    pack_b_kernel<<<(H_ * H_ + 255) / 256, 256, 0, stream>>>(W_rec, WrecP, H_, H_);
    pack_b_kernel<<<(O_ * H_ + 255) / 256, 256, 0, stream>>>(W_out, WoutP, H_, O_);

    xproj_kernel<<<(8192 * 32) / 8, 256, 0, stream>>>(inputs, WinP, b_in, xp);
    rnn_scan_kernel<<<4, 512, 0, stream>>>(xp, WrecP, h_bias, hs);
    out_kernel<<<(8192 * 4) / 8, 256, 0, stream>>>(hs, WoutP, b_out, out);
}
